// S4Block_18777597018518
// MI455X (gfx1250) — matrix operand order repaired and checked by reading, not compiled, not run on hardware
//
#include <hip/hip_runtime.h>
#include <hip/hip_bf16.h>

// ---------------------------------------------------------------------------
// S4 block on MI455X (gfx1250, wave32, WMMA, TDM).
//   xB = x @ B^T          : (16384,2048)x(2048,128)   bf16 WMMA, f32 accum
//   h_t = tanh(h A^T + xB): 2048-step latency chain, 1 WGP, WMMA per step
//   y  = hs @ C^T + x*D   : (16384,128)x(128,2048)    bf16 WMMA, fused skip
// HBM traffic ~400MB -> ~17us floor at 23.3 TB/s; bf16 WMMA keeps the GEMMs
// on the memory roofline (f32 16x16x4 WMMA would be compute bound).
// proj_in: b128 global loads + ping-pong LDS (1 barrier / K-iter).
// proj_out: hs tile staged by the Tensor Data Mover (TENSORcnt) when the
//           builtin is available; manual copy fallback otherwise.
// ---------------------------------------------------------------------------

#define BATCH   8
#define SEQLEN  2048
#define DMODEL  2048
#define DSTATE  128
#define MTOT    (BATCH * SEQLEN)   // 16384 GEMM rows

typedef __bf16        bf16x16 __attribute__((ext_vector_type(16)));
typedef float         floatx8 __attribute__((ext_vector_type(8)));
typedef unsigned int  uintx4  __attribute__((ext_vector_type(4)));
typedef int           intx4   __attribute__((ext_vector_type(4)));
typedef int           intx8   __attribute__((ext_vector_type(8)));

union Frag { unsigned int u[8]; bf16x16 v; };

#if defined(__has_builtin)
#if __has_builtin(__builtin_amdgcn_tensor_load_to_lds)
#define HAVE_TDM 1
#endif
#endif

__device__ __forceinline__ unsigned int pack_bf16(float lo, float hi) {
    union { __bf16 h; unsigned short s; } a, b;
    a.h = (__bf16)lo;
    b.h = (__bf16)hi;
    return (unsigned int)a.s | ((unsigned int)b.s << 16);
}

// A-operand (16x32 bf16) pair-slot for VGPR v, lane-group g (ISA 7.12.2):
//   K = 16*(v>>2) + 8*g + 2*(v&3)
__device__ __forceinline__ int a_pair(int v, int g) {
    return 8 * (v >> 2) + 4 * g + (v & 3);
}
// B-operand (32x16 bf16) pair-slot: K = 16*(v>>2) + 8*g + 2*(v&3)
__device__ __forceinline__ int b_pair(int v, int g) {
    return 8 * (v >> 2) + 4 * g + (v & 3);
}

// ---------------------------------------------------------------------------
// Kernel 1: xB[m][s] = sum_d x[m][d] * B[s][d]      (m = b*SEQLEN + l)
// Block: 128 threads (4 waves). Tile: M=64 (16/wave), N=128 (full d_state).
// Ping-pong LDS double buffer: one barrier per K-iter, loads overlap WMMA.
// ---------------------------------------------------------------------------
__global__ void __launch_bounds__(128)
s4_proj_in(const float* __restrict__ x, const float* __restrict__ Bm,
           float* __restrict__ xB) {
    const int tid  = threadIdx.x;
    const int lane = tid & 31;
    const int wv   = tid >> 5;          // 0..3
    const int g    = lane >> 4;         // lane group
    const int lm   = lane & 15;
    const int m0   = blockIdx.x * 64;

    __shared__ unsigned int ldsX[2][64][17];    // 64 x 16 bf16-pairs (+pad)
    __shared__ unsigned int ldsB[2][128][17];   // 128 x 16 pairs (+pad)

    const float4* x4 = (const float4*)x;
    const float4* B4 = (const float4*)Bm;

    auto stage = [&](int kc, int buf) {
        // x tile: 64 rows x 8 float4 = 512 slots / 128 threads
#pragma unroll
        for (int i = 0; i < 4; ++i) {
            int slot = tid + 128 * i;
            int m = slot >> 3, q = slot & 7;
            float4 v = x4[(size_t)(m0 + m) * (DMODEL / 4) + (kc >> 2) + q];
            ldsX[buf][m][2 * q]     = pack_bf16(v.x, v.y);
            ldsX[buf][m][2 * q + 1] = pack_bf16(v.z, v.w);
        }
        // B tile: 128 rows x 8 float4 = 1024 slots
#pragma unroll
        for (int i = 0; i < 8; ++i) {
            int slot = tid + 128 * i;
            int n = slot >> 3, q = slot & 7;
            float4 v = B4[(size_t)n * (DMODEL / 4) + (kc >> 2) + q];
            ldsB[buf][n][2 * q]     = pack_bf16(v.x, v.y);
            ldsB[buf][n][2 * q + 1] = pack_bf16(v.z, v.w);
        }
    };

    floatx8 acc[8];
#pragma unroll
    for (int n = 0; n < 8; ++n) { floatx8 z = {}; acc[n] = z; }

    stage(0, 0);
    __syncthreads();

    for (int kc = 0; kc < DMODEL; kc += 32) {
        const int cur = (kc >> 5) & 1;
        if (kc + 32 < DMODEL)
            stage(kc + 32, cur ^ 1);    // fill other buffer while computing

        Frag a;
#pragma unroll
        for (int v = 0; v < 8; ++v)
            a.u[v] = ldsX[cur][wv * 16 + lm][a_pair(v, g)];

#pragma unroll
        for (int n = 0; n < 8; ++n) {
            Frag b;
#pragma unroll
            for (int v = 0; v < 8; ++v)
                b.u[v] = ldsB[cur][n * 16 + lm][b_pair(v, g)];
            acc[n] = __builtin_amdgcn_wmma_f32_16x16x32_bf16(
                false, a.v, false, b.v, (short)0, acc[n], false, false);
        }
        __syncthreads();   // next buffer staged & this buffer's reads done
    }

    // D-tile layout: VGPR r, lanes0-15 -> M=r, lanes16-31 -> M=8+r; N=lane&15
#pragma unroll
    for (int n = 0; n < 8; ++n)
#pragma unroll
        for (int r = 0; r < 8; ++r) {
            int ml = wv * 16 + r + 8 * g;
            int nl = n * 16 + lm;
            xB[(size_t)(m0 + ml) * DSTATE + nl] = acc[n][r];
        }
}

// ---------------------------------------------------------------------------
// Kernel 2: recurrence. One workgroup, 8 waves. State H padded 8->16 rows.
// Wave w owns state columns [16w, 16w+16). A weights preloaded as per-wave
// B-operand fragments; per step: 4 WMMAs + tanh + LDS exchange (2 barriers).
// hs written directly as bf16 for the output-projection GEMM.
// ---------------------------------------------------------------------------
__global__ void __launch_bounds__(256)
s4_recurrence(const float* __restrict__ Am, const float* __restrict__ xB,
              __bf16* __restrict__ hs) {
    const int tid  = threadIdx.x;
    const int lane = tid & 31;
    const int wv   = tid >> 5;          // 0..7
    const int g    = lane >> 4;
    const int lm   = lane & 15;
    const int col  = 16 * wv + lm;      // this lane's state column

    __shared__ __bf16 H[16][136];       // 16 (padded batch) x 128 state (+pad)

    for (int i = tid; i < 16 * 136; i += 256)
        (&H[0][0])[i] = (__bf16)0.0f;

    // Preload A^T fragments: B-operand value (K=k, n) = A[16w+n][k]
    Frag bfr[4];
#pragma unroll
    for (int c = 0; c < 4; ++c)
#pragma unroll
        for (int v = 0; v < 8; ++v) {
            int row = 16 * wv + lm;
            int k   = c * 32 + 16 * (v >> 2) + 8 * g + 2 * (v & 3);
            bfr[c].u[v] = pack_bf16(Am[row * DSTATE + k],
                                    Am[row * DSTATE + k + 1]);
        }
    __syncthreads();

    for (int t = 0; t < SEQLEN; ++t) {
        floatx8 c = {};
#pragma unroll
        for (int kc = 0; kc < 4; ++kc) {
            Frag a;
#pragma unroll
            for (int v = 0; v < 8; ++v) {
                int k = kc * 32 + 2 * a_pair(v, g);
                a.u[v] = *(const unsigned int*)&H[lm][k];
            }
            c = __builtin_amdgcn_wmma_f32_16x16x32_bf16(
                false, a.v, false, bfr[kc].v, (short)0, c, false, false);
        }
        __syncthreads();   // all waves done reading old H
#pragma unroll
        for (int r = 0; r < 8; ++r) {
            float val = c[r];
            if (g == 0)    // real batches live in rows 0..7 (lane group 0)
                val += xB[((size_t)r * SEQLEN + t) * DSTATE + col];
            val = tanhf(val);
            __bf16 hb = (__bf16)val;
            H[r + 8 * g][col] = hb;     // rows 8..15: pad = tanh(0) = 0
            if (g == 0)
                hs[((size_t)r * SEQLEN + t) * DSTATE + col] = hb;
        }
        __syncthreads();   // new H visible to every wave
    }
}

// ---------------------------------------------------------------------------
// Kernel 3: y[m][d] = sum_s hs[m][s]*C[d][s] + x[m][d]*D[d]
// Block: 256 threads (8 waves as 2x4). Tile: M=64, N=128, K=128 (staged once).
// hs tile (bf16, no conversion needed) staged by TDM when available.
// ---------------------------------------------------------------------------
__global__ void __launch_bounds__(256)
s4_proj_out(const __bf16* __restrict__ hs, const float* __restrict__ Cm,
            const float* __restrict__ x, const float* __restrict__ Dv,
            float* __restrict__ y) {
    const int tid  = threadIdx.x;
    const int lane = tid & 31;
    const int wv   = tid >> 5;
    const int g    = lane >> 4;
    const int lm   = lane & 15;
    const int wm   = wv >> 2;           // 0..1  (M dir)
    const int wn   = wv & 3;            // 0..3  (N dir)
    const int m0   = blockIdx.x * 64;
    const int n0   = blockIdx.y * 128;

    __shared__ unsigned int ldsH[64][65];    // 64 rows x 64 bf16-pairs (+pad)
    __shared__ unsigned int ldsC[128][65];   // 128 rows x 64 pairs (+pad)

#if HAVE_TDM
    // ---- Tensor Data Mover: 64x128 bf16 tile, LDS pad 1 DW per 64 DW ----
    if (wv == 0) {
        unsigned long long ga =
            (unsigned long long)(const void*)(hs + (size_t)m0 * DSTATE);
        unsigned int ldsa =
            (unsigned int)(unsigned long long)(void*)&ldsH[0][0];

        uintx4 g0;
        g0[0] = 1u;                                    // count=1 (valid D#)
        g0[1] = ldsa;                                  // lds_addr
        g0[2] = (unsigned int)(ga & 0xFFFFFFFFull);    // global_addr[31:0]
        g0[3] = (unsigned int)((ga >> 32) & 0x1FFFFFFull) | (2u << 30); // +type=2

        intx8 g1;
        g1[0] = (int)((1u << 16)        // data_size = 2 bytes
                    | (1u << 20)        // pad_enable
                    | (5u << 22));      // pad_interval: 64 DWORDs, amount: 1 DW
        g1[1] = (int)(128u << 16);      // tensor_dim0 = 128 (low 16)
        g1[2] = (int)(((unsigned)MTOT & 0xFFFFu) << 16); // dim0 hi=0 | dim1 lo
        g1[3] = (int)(128u << 16);      // dim1 hi = 0 | tile_dim0 = 128
        g1[4] = 64;                     // tile_dim1 = 64, tile_dim2 = 0
        g1[5] = 128;                    // tensor_dim0_stride = 128 (low 32)
        g1[6] = 0;                      // stride hi | dim1_stride lo
        g1[7] = 0;

        intx4 g2 = (intx4)0;
        intx4 g3 = (intx4)0;
#if __clang_major__ >= 23
        intx8 g4 = (intx8)0;
        __builtin_amdgcn_tensor_load_to_lds(g0, g1, g2, g3, g4, 0);
#else
        __builtin_amdgcn_tensor_load_to_lds(g0, g1, g2, g3, 0);
#endif
        __builtin_amdgcn_s_wait_tensorcnt(0);
    }
#else
    // ---- fallback: manual pair copy (hs already bf16 in memory) ----
    const unsigned int* hsP = (const unsigned int*)hs;
#pragma unroll
    for (int i = 0; i < 16; ++i) {
        int slot = tid + 256 * i;
        int ml = slot >> 6, p = slot & 63;
        ldsH[ml][p] = hsP[(size_t)(m0 + ml) * (DSTATE / 2) + p];
    }
#endif

    const float2* C2 = (const float2*)Cm;
#pragma unroll
    for (int i = 0; i < 32; ++i) {
        int slot = tid + 256 * i;
        int nl = slot >> 6, p = slot & 63;
        float2 cv = C2[(size_t)(n0 + nl) * (DSTATE / 2) + p];
        ldsC[nl][p] = pack_bf16(cv.x, cv.y);
    }
    __syncthreads();

    floatx8 acc[2][2];
#pragma unroll
    for (int i = 0; i < 2; ++i)
#pragma unroll
        for (int j = 0; j < 2; ++j) { floatx8 z = {}; acc[i][j] = z; }

#pragma unroll
    for (int kc = 0; kc < 4; ++kc) {
        Frag a[2], b[2];
#pragma unroll
        for (int i = 0; i < 2; ++i)
#pragma unroll
            for (int v = 0; v < 8; ++v)
                a[i].u[v] = ldsH[wm * 32 + i * 16 + lm][kc * 16 + a_pair(v, g)];
#pragma unroll
        for (int j = 0; j < 2; ++j)
#pragma unroll
            for (int v = 0; v < 8; ++v)
                b[j].u[v] = ldsC[wn * 32 + j * 16 + lm][kc * 16 + b_pair(v, g)];
#pragma unroll
        for (int i = 0; i < 2; ++i)
#pragma unroll
            for (int j = 0; j < 2; ++j)
                acc[i][j] = __builtin_amdgcn_wmma_f32_16x16x32_bf16(
                    false, a[i].v, false, b[j].v, (short)0, acc[i][j],
                    false, false);
    }

    // fused epilogue: y = acc + x*D (single pass over x / y)
#pragma unroll
    for (int i = 0; i < 2; ++i)
#pragma unroll
        for (int j = 0; j < 2; ++j)
#pragma unroll
            for (int r = 0; r < 8; ++r) {
                int ml = wm * 32 + i * 16 + r + 8 * g;
                int nl = wn * 32 + j * 16 + lm;
                size_t idx = (size_t)(m0 + ml) * DMODEL + (n0 + nl);
                y[idx] = acc[i][j][r] + x[idx] * Dv[n0 + nl];
            }
}

// ---------------------------------------------------------------------------
extern "C" void kernel_launch(void* const* d_in, const int* in_sizes, int n_in,
                              void* d_out, int out_size, void* d_ws,
                              size_t ws_size, hipStream_t stream) {
    const float* x  = (const float*)d_in[0];  // (8, 2048, 2048)
    const float* Am = (const float*)d_in[1];  // (128, 128)
    const float* Bm = (const float*)d_in[2];  // (128, 2048)
    const float* Cm = (const float*)d_in[3];  // (2048, 128)
    const float* Dv = (const float*)d_in[4];  // (2048,)
    float* y = (float*)d_out;

    float*  xB = (float*)d_ws;                               // 8 MB f32
    __bf16* hs = (__bf16*)((char*)d_ws +
                           (size_t)MTOT * DSTATE * sizeof(float)); // 4 MB bf16

    s4_proj_in<<<dim3(MTOT / 64), dim3(128), 0, stream>>>(x, Bm, xB);
    s4_recurrence<<<dim3(1), dim3(256), 0, stream>>>(Am, xB, hs);
    s4_proj_out<<<dim3(MTOT / 64, DMODEL / 128), dim3(256), 0, stream>>>(
        hs, Cm, x, Dv, y);
}
